// MultiHeadExternalAttention_83906481094719
// MI455X (gfx1250) — matrix operand order repaired and checked by reading, not compiled, not run on hardware
//
#include <hip/hip_runtime.h>
#include <hip/hip_bf16.h>
#include <stdint.h>

// ---------------------------------------------------------------------------
// MultiHeadExternalAttention for MI455X (gfx1250, wave32, WMMA bf16)
//
// Pre-pass converts x / trans_w / proj_w / lin0_w / lin1_w to bf16 once, so
// every GEMM stage loop is a pure GLOBAL_LOAD_ASYNC_TO_LDS_B128 (ASYNCcnt),
// double-buffered, feeding ds_load_b128 -> v_wmma_f32_16x16x32_bf16.
//
// Workspace layout (d_ws), ~583 MB:
//   [0)           bf16 logits [B][H][N][64]     268,435,456 B
//   [268435456)   bf16 o      [B][N][4096]      268,435,456 B
//   [536870912)   f32  partial[B*H][64blk][64]    4,194,304 B
//   [541065216)   f32  colsum [B*H][64]              65,536 B
//   [541130752)   bf16 x_bf   [B][N][512]        33,554,432 B
//   [574685184)   bf16 tw_bf  [4096][512]         4,194,304 B
//   [578879488)   bf16 pw_bf  [512][4096]         4,194,304 B
//   [583073792)   bf16 l0_bf  [64][64]                8,192 B
//   [583081984)   bf16 l1_bf  [64][64]                8,192 B
// ---------------------------------------------------------------------------

typedef __bf16 bf16_t;
typedef __bf16 v16bf __attribute__((ext_vector_type(16)));
typedef __bf16 v8bf  __attribute__((ext_vector_type(8)));
typedef __bf16 v4bf  __attribute__((ext_vector_type(4)));
typedef float  v8f   __attribute__((ext_vector_type(8)));

#define BATCH   4
#define NSEQ    8192
#define DIMC    512
#define HD      4096
#define HEADS   64
#define NBLK1   (NSEQ / 128)

// ---- fragment gathers (16B-aligned; LDS ptr -> ds_load_b128) --------------
__device__ __forceinline__ v16bf fragA(const bf16_t* rowptr, int half) {
  v8bf lo = *(const v8bf*)(rowptr + half * 8);
  v8bf hi = *(const v8bf*)(rowptr + 16 + half * 8);
  v16bf a;
#pragma unroll
  for (int i = 0; i < 8; ++i) { a[i] = lo[i]; a[i + 8] = hi[i]; }
  return a;
}
__device__ __forceinline__ v16bf fragB(const bf16_t* rowptr, int half) {
  v8bf lo = *(const v8bf*)(rowptr + half * 8);
  v8bf hi = *(const v8bf*)(rowptr + 16 + half * 8);
  v16bf b;
#pragma unroll
  for (int i = 0; i < 8; ++i) { b[i] = lo[i]; b[i + 8] = hi[i]; }
  return b;
}
__device__ __forceinline__ v4bf cvt4(float4 v) {
  v4bf r;
  r[0] = (bf16_t)v.x; r[1] = (bf16_t)v.y; r[2] = (bf16_t)v.z; r[3] = (bf16_t)v.w;
  return r;
}
// CDNA5 async copy: global memory -> LDS, 16 bytes per lane, ASYNCcnt-tracked.
__device__ __forceinline__ void async_copy_b128(uint32_t lds_off, const void* gptr) {
  asm volatile("global_load_async_to_lds_b128 %0, %1, off"
               :: "v"(lds_off), "v"(gptr) : "memory");
}
__device__ __forceinline__ void wait_asynccnt0() {
  asm volatile("s_wait_asynccnt 0x0" ::: "memory");
}
__device__ __forceinline__ uint32_t lds_off(const void* p) {
  return (uint32_t)(uintptr_t)p;   // generic LDS addr: low 32 bits = LDS offset
}

// ===========================================================================
// Pre-pass: f32 -> bf16 (vectorized), removes all cvt work from GEMM loops
// ===========================================================================
__global__ __launch_bounds__(256)
void k_cvt(const float* __restrict__ src, bf16_t* __restrict__ dst, int n4) {
  int i = blockIdx.x * blockDim.x + threadIdx.x;
  if (i < n4)
    *(v4bf*)&dst[(size_t)i * 4] = cvt4(*(const float4*)&src[(size_t)i * 4]);
}

// ===========================================================================
// Kernel 1: h = x * trans_w^T + trans_b ; logit = h_head * lin0_w^T + lin0_b
// 2 heads per block (halves x/trans_w L2 re-reads); async double-buffered.
// grid = (N/128, HEADS/2, B), block = 256 (8 waves); wave: 16 rows
// ===========================================================================
__global__ __launch_bounds__(256)
void k_trans_lin0(const bf16_t* __restrict__ xb,
                  const bf16_t* __restrict__ twb,
                  const float* __restrict__ trans_b,
                  const bf16_t* __restrict__ l0b,
                  const float* __restrict__ lin0_b,
                  bf16_t* __restrict__ logits,
                  float* __restrict__ partial) {
  __shared__ __attribute__((aligned(16))) bf16_t xs[2][128][40];
  __shared__ __attribute__((aligned(16))) bf16_t wl[2][128][40];
  __shared__ __attribute__((aligned(16))) bf16_t hs[128][72];
  __shared__ float psum[8][64];

  const int tid  = threadIdx.x;
  const int wave = tid >> 5;
  const int lane = tid & 31;
  const int half = lane >> 4;
  const int lm   = lane & 15;

  const int nblk  = blockIdx.x;
  const int head0 = blockIdx.y * 2;
  const int b     = blockIdx.z;
  const int n0    = nblk * 128;
  const int d0    = head0 * 64;

  // async stage one K-slice: x rows (128x32) + trans_w rows for 2 heads (128x32)
  auto stage = [&](int buf, int kk) {
#pragma unroll
    for (int p = 0; p < 2; ++p) {
      int idx = p * 256 + tid;                 // 0..511
      int r = idx >> 2, kq = (idx & 3) * 8;
      async_copy_b128(lds_off(&xs[buf][r][kq]),
                      &xb[((size_t)b * NSEQ + n0 + r) * DIMC + kk + kq]);
      async_copy_b128(lds_off(&wl[buf][r][kq]),
                      &twb[(size_t)(d0 + r) * DIMC + kk + kq]);
    }
  };
  stage(0, 0);

  v8f acc[2][4] = {};
  int cur = 0;
  for (int kk = 0; kk < DIMC; kk += 32, cur ^= 1) {
    wait_asynccnt0();        // our async writes into buf[cur] landed in LDS
    __syncthreads();         // all waves' writes visible
    if (kk + 32 < DIMC) stage(cur ^ 1, kk + 32);
    v16bf a = fragA(&xs[cur][wave * 16 + lm][0], half);
#pragma unroll
    for (int hh = 0; hh < 2; ++hh)
#pragma unroll
      for (int t = 0; t < 4; ++t) {
        v16bf bb = fragB(&wl[cur][hh * 64 + t * 16 + lm][0], half);
        acc[hh][t] = __builtin_amdgcn_wmma_f32_16x16x32_bf16(
            false, a, false, bb, (short)0, acc[hh][t], false, false);
      }
  }

  // lin0 B-fragments (head-independent) straight from global bf16 -> registers
  v16bf bfr[2][4];
#pragma unroll
  for (int ks2 = 0; ks2 < 2; ++ks2)
#pragma unroll
    for (int t = 0; t < 4; ++t)
      bfr[ks2][t] = fragB(&l0b[(t * 16 + lm) * 64 + ks2 * 32], half);

  for (int hh = 0; hh < 2; ++hh) {
    const int head = head0 + hh;
    const int dh   = head * 64;
    __syncthreads();                 // previous head's hs consumers are done
    // stage h tile (bias + bf16 round)
#pragma unroll
    for (int t = 0; t < 4; ++t) {
      int n = t * 16 + lm;
      float bias = trans_b[dh + n];
#pragma unroll
      for (int r = 0; r < 8; ++r)
        hs[wave * 16 + half * 8 + r][n] = (bf16_t)(acc[hh][t][r] + bias);
    }
    __syncthreads();
    // logit = h (128x64) * lin0^T, K=64
    v8f e[4] = {};
#pragma unroll
    for (int ks2 = 0; ks2 < 2; ++ks2) {
      v16bf a = fragA(&hs[wave * 16 + lm][ks2 * 32], half);
#pragma unroll
      for (int t = 0; t < 4; ++t)
        e[t] = __builtin_amdgcn_wmma_f32_16x16x32_bf16(
            false, a, false, bfr[ks2][t], (short)0, e[t], false, false);
    }
    __syncthreads();                 // everyone done reading hs
    // bias, bf16 round, exp column partials (deterministic order)
    float csum[4];
#pragma unroll
    for (int t = 0; t < 4; ++t) {
      int j = t * 16 + lm;
      float bias = lin0_b[j];
      float s = 0.f;
#pragma unroll
      for (int r = 0; r < 8; ++r) {
        bf16_t Lb = (bf16_t)(e[t][r] + bias);
        hs[wave * 16 + half * 8 + r][j] = Lb;
        s += __expf((float)Lb);
      }
      s += __shfl_xor(s, 16);
      csum[t] = s;
    }
    if (lane < 16) {
#pragma unroll
      for (int t = 0; t < 4; ++t) psum[wave][t * 16 + lane] = csum[t];
    }
    __syncthreads();
    // coalesced logit store (128-bit LDS reads -> 128-bit global stores)
#pragma unroll
    for (int p = 0; p < 4; ++p) {
      int idx = p * 256 + tid;               // 0..1023
      int r = idx >> 3, j8 = (idx & 7) * 8;
      *(v8bf*)&logits[(((size_t)b * HEADS + head) * NSEQ + n0 + r) * 64 + j8] =
          *(const v8bf*)&hs[r][j8];
    }
    if (tid < 64) {
      float s = 0.f;
#pragma unroll
      for (int w = 0; w < 8; ++w) s += psum[w][tid];
      partial[(((size_t)b * HEADS + head) * NBLK1 + nblk) * 64 + tid] = s;
    }
  }
}

// ===========================================================================
// Kernel 2: colsum[b,h,j] = ordered sum over the 64 n-blocks (deterministic)
// ===========================================================================
__global__ __launch_bounds__(256)
void k_colsum(const float* __restrict__ partial, float* __restrict__ colsum) {
  int idx = blockIdx.x * blockDim.x + threadIdx.x;
  if (idx >= BATCH * HEADS * 64) return;
  int j = idx & 63;
  int bh = idx >> 6;
  float s = 0.f;
  for (int blk = 0; blk < NBLK1; ++blk)
    s += partial[((size_t)bh * NBLK1 + blk) * 64 + j];
  colsum[idx] = s;
}

// ===========================================================================
// Kernel 3: softmax over n (via colsum), row double-normalization, * lin1^T
// Logits tile and lin1 weights async-loaded to LDS.
// grid = (N/64, HEADS, B), block = 128 (4 waves)
// ===========================================================================
__global__ __launch_bounds__(128)
void k_softmax_lin1(const bf16_t* __restrict__ logits,
                    const float* __restrict__ colsum,
                    const bf16_t* __restrict__ l1b,
                    const float* __restrict__ lin1_b,
                    bf16_t* __restrict__ o) {
  __shared__ __attribute__((aligned(16))) float  af[64][68];
  __shared__ __attribute__((aligned(16))) bf16_t at[64][72];
  __shared__ __attribute__((aligned(16))) bf16_t l1[64][72];
  __shared__ float rcs[64];
  __shared__ float rs[64][2];

  const int tid  = threadIdx.x;
  const int wave = tid >> 5;
  const int lane = tid & 31;
  const int half = lane >> 4;
  const int lm   = lane & 15;

  const int n0   = blockIdx.x * 64;
  const int head = blockIdx.y;
  const int b    = blockIdx.z;

  // async: logits tile (64x64) -> at, lin1 (64x64) -> l1
#pragma unroll
  for (int p = 0; p < 4; ++p) {
    int idx = p * 128 + tid;                   // 0..511
    int r = idx >> 3, kq = (idx & 7) * 8;
    async_copy_b128(lds_off(&at[r][kq]),
        &logits[(((size_t)b * HEADS + head) * NSEQ + n0 + r) * 64 + kq]);
    async_copy_b128(lds_off(&l1[r][kq]), &l1b[r * 64 + kq]);
  }
  if (tid < 64)
    rcs[tid] = 1.0f / colsum[((size_t)b * HEADS + head) * 64 + tid];
  wait_asynccnt0();
  __syncthreads();

  // exp / colsum + partial row sums (2 threads/row, 32 cols, ds b128 reads)
  {
    int row = tid >> 1, ch = tid & 1;
    float s = 0.f;
#pragma unroll
    for (int q = 0; q < 4; ++q) {
      v8bf v = *(const v8bf*)&at[row][ch * 32 + q * 8];
#pragma unroll
      for (int k = 0; k < 8; ++k) {
        int c = ch * 32 + q * 8 + k;
        float e = __expf((float)v[k]) * rcs[c];
        af[row][c] = e;
        s += e;
      }
    }
    rs[row][ch] = s;
  }
  __syncthreads();
  {
    int row = tid >> 1, ch = tid & 1;
    float inv = 1.0f / (1e-10f + rs[row][0] + rs[row][1]);
#pragma unroll
    for (int q = 0; q < 8; ++q) {
      float4 v = *(const float4*)&af[row][ch * 32 + q * 4];
      v.x *= inv; v.y *= inv; v.z *= inv; v.w *= inv;
      *(v4bf*)&at[row][ch * 32 + q * 4] = cvt4(v);
    }
  }
  __syncthreads();

  // o_tile = attn (64x64) * lin1^T, K=64
  v8f c[4] = {};
#pragma unroll
  for (int ks = 0; ks < 64; ks += 32) {
    v16bf a = fragA(&at[wave * 16 + lm][ks], half);
#pragma unroll
    for (int t = 0; t < 4; ++t) {
      v16bf bb = fragB(&l1[t * 16 + lm][ks], half);
      c[t] = __builtin_amdgcn_wmma_f32_16x16x32_bf16(
          false, a, false, bb, (short)0, c[t], false, false);
    }
  }
  __syncthreads();  // all waves finished reading at

  // +bias, restage into LDS, then coalesced 128-bit stores
#pragma unroll
  for (int t = 0; t < 4; ++t) {
    int j = t * 16 + lm;
    float bias = lin1_b[j];
#pragma unroll
    for (int r = 0; r < 8; ++r)
      at[wave * 16 + half * 8 + r][j] = (bf16_t)(c[t][r] + bias);
  }
  __syncthreads();
  {
    int row = tid >> 1, ch = tid & 1;
    v8bf* dst = (v8bf*)&o[((size_t)b * NSEQ + n0 + row) * HD + head * 64 + ch * 32];
#pragma unroll
    for (int q = 0; q < 4; ++q) dst[q] = *(const v8bf*)&at[row][ch * 32 + q * 8];
  }
}

// ===========================================================================
// Kernel 4: out = o (bf16) * proj_w^T + proj_b  (f32 out)
// Both operands async-staged, double-buffered LDS.
// grid = (B*N/128, 512/64), block = 256 (8 waves)
// ===========================================================================
__global__ __launch_bounds__(256)
void k_proj(const bf16_t* __restrict__ o,
            const bf16_t* __restrict__ pwb,
            const float* __restrict__ proj_b,
            float* __restrict__ out) {
  __shared__ __attribute__((aligned(16))) bf16_t os[2][128][40];
  __shared__ __attribute__((aligned(16))) bf16_t pw[2][64][40];

  const int tid  = threadIdx.x;
  const int wave = tid >> 5;
  const int lane = tid & 31;
  const int half = lane >> 4;
  const int lm   = lane & 15;

  const size_t row0 = (size_t)blockIdx.x * 128;
  const int    col0 = blockIdx.y * 64;

  auto stage = [&](int buf, int kk) {
#pragma unroll
    for (int p = 0; p < 2; ++p) {
      int idx = p * 256 + tid;                 // 0..511
      int r = idx >> 2, kq = (idx & 3) * 8;
      async_copy_b128(lds_off(&os[buf][r][kq]), &o[(row0 + r) * HD + kk + kq]);
    }
    {
      int r = tid >> 2, kq = (tid & 3) * 8;    // 64 rows x 4 ops
      async_copy_b128(lds_off(&pw[buf][r][kq]),
                      &pwb[(size_t)(col0 + r) * HD + kk + kq]);
    }
  };
  stage(0, 0);

  v8f c[4] = {};
  int cur = 0;
  for (int kk = 0; kk < HD; kk += 32, cur ^= 1) {
    wait_asynccnt0();
    __syncthreads();
    if (kk + 32 < HD) stage(cur ^ 1, kk + 32);
    v16bf a = fragA(&os[cur][wave * 16 + lm][0], half);
#pragma unroll
    for (int t = 0; t < 4; ++t) {
      v16bf bb = fragB(&pw[cur][t * 16 + lm][0], half);
      c[t] = __builtin_amdgcn_wmma_f32_16x16x32_bf16(
          false, a, false, bb, (short)0, c[t], false, false);
    }
  }

#pragma unroll
  for (int t = 0; t < 4; ++t) {
    int n = col0 + t * 16 + lm;
    float bias = proj_b[n];
#pragma unroll
    for (int r = 0; r < 8; ++r) {
      size_t row = row0 + wave * 16 + half * 8 + r;
      out[row * DIMC + n] = c[t][r] + bias;
    }
  }
}

// ===========================================================================
extern "C" void kernel_launch(void* const* d_in, const int* in_sizes, int n_in,
                              void* d_out, int out_size, void* d_ws,
                              size_t ws_size, hipStream_t stream) {
  const float* x       = (const float*)d_in[0];
  const float* trans_w = (const float*)d_in[1];
  const float* trans_b = (const float*)d_in[2];
  const float* lin0_w  = (const float*)d_in[3];
  const float* lin0_b  = (const float*)d_in[4];
  const float* lin1_w  = (const float*)d_in[5];
  const float* lin1_b  = (const float*)d_in[6];
  const float* proj_w  = (const float*)d_in[7];
  const float* proj_b  = (const float*)d_in[8];
  float* out = (float*)d_out;

  char* ws = (char*)d_ws;
  bf16_t* logits  = (bf16_t*)ws;                          // 268,435,456 B
  bf16_t* o       = (bf16_t*)(ws + 268435456ULL);         // 268,435,456 B
  float*  partial = (float*)(ws + 536870912ULL);          //   4,194,304 B
  float*  colsum  = (float*)(ws + 541065216ULL);          //      65,536 B
  bf16_t* xb      = (bf16_t*)(ws + 541130752ULL);         //  33,554,432 B
  bf16_t* twb     = (bf16_t*)(ws + 574685184ULL);         //   4,194,304 B
  bf16_t* pwb     = (bf16_t*)(ws + 578879488ULL);         //   4,194,304 B
  bf16_t* l0b     = (bf16_t*)(ws + 583073792ULL);         //       8,192 B
  bf16_t* l1b     = (bf16_t*)(ws + 583081984ULL);         //       8,192 B

  // pre-pass: bf16 conversions (one-time, ~126 MB of traffic)
  k_cvt<<<(BATCH * NSEQ * DIMC / 4 + 255) / 256, 256, 0, stream>>>(x, xb,
      BATCH * NSEQ * DIMC / 4);
  k_cvt<<<(HD * DIMC / 4 + 255) / 256, 256, 0, stream>>>(trans_w, twb,
      HD * DIMC / 4);
  k_cvt<<<(DIMC * HD / 4 + 255) / 256, 256, 0, stream>>>(proj_w, pwb,
      DIMC * HD / 4);
  k_cvt<<<4, 256, 0, stream>>>(lin0_w, l0b, 64 * 64 / 4);
  k_cvt<<<4, 256, 0, stream>>>(lin1_w, l1b, 64 * 64 / 4);

  dim3 g1(NSEQ / 128, HEADS / 2, BATCH);
  k_trans_lin0<<<g1, 256, 0, stream>>>(xb, twb, trans_b, l0b, lin0_b,
                                       logits, partial);

  k_colsum<<<(BATCH * HEADS * 64 + 255) / 256, 256, 0, stream>>>(partial,
                                                                 colsum);

  dim3 g2(NSEQ / 64, HEADS, BATCH);
  k_softmax_lin1<<<g2, 128, 0, stream>>>(logits, colsum, l1b, lin1_b, o);

  dim3 g3((BATCH * NSEQ) / 128, DIMC / 64, 1);
  k_proj<<<g3, 256, 0, stream>>>(o, pwb, proj_b, out);
}
